// MixedChunkAttention_64226940945008
// MI455X (gfx1250) — matrix operand order repaired and checked by reading, not compiled, not run on hardware
//
#include <hip/hip_runtime.h>
#include <cmath>

// ---------------- problem constants ----------------
#define B_SZ   2
#define N_SEQ  8190
#define DIMX   512
#define QKD    128
#define HIDX   1024
#define G_SZ   21
#define GCHUNK 390
#define BN     (B_SZ * N_SEQ)      // 16380
#define MAXREL 32
#define EPSF   1e-5f
#define KV_SPLITS 16
#define LDP    40                  // padded LDS row stride (bf16 elems), 80B, 16B-aligned

typedef __attribute__((ext_vector_type(16))) __bf16 v16bf;
typedef __attribute__((ext_vector_type(8)))  __bf16 v8bf;
typedef __attribute__((ext_vector_type(8)))  float  v8f;

__device__ __forceinline__ float siluf(float x) {
    return x / (1.f + __expf(-x));
}

__device__ __forceinline__ v16bf frag_cat(v8bf lo, v8bf hi) {
    return __builtin_shufflevector(lo, hi, 0, 1, 2, 3, 4, 5, 6, 7,
                                   8, 9, 10, 11, 12, 13, 14, 15);
}

// A-fragment (16x32 bf16, MxK): lane lr = row, runs k=[lh*8,+8) and [16+lh*8,+8)
__device__ __forceinline__ v16bf load_afrag(const __bf16* rowp, int lh) {
    v8bf lo = *(const v8bf*)(rowp + lh * 8);
    v8bf hi = *(const v8bf*)(rowp + 16 + lh * 8);
    return frag_cat(lo, hi);
}
// B-fragment (32x16 bf16, KxN) from transposed LDS [col][k]: runs k=[lh*8,+8) and [16+lh*8,+8)
__device__ __forceinline__ v16bf load_bfrag(const __bf16* colp, int lh) {
    v8bf lo = *(const v8bf*)(colp + lh * 8);
    v8bf hi = *(const v8bf*)(colp + 16 + lh * 8);
    return frag_cat(lo, hi);
}

// ---------------- fp32 -> bf16 cast ----------------
__global__ __launch_bounds__(256) void f2bf_kernel(const float* __restrict__ in,
                                                   __bf16* __restrict__ out, int n) {
    for (int i = blockIdx.x * blockDim.x + threadIdx.x; i < n; i += gridDim.x * blockDim.x)
        out[i] = (__bf16)in[i];
}

// ---------------- pre layernorm: x (BN,512) -> xn bf16 ----------------
__global__ __launch_bounds__(256) void ln_pre_kernel(const float* __restrict__ x,
                                                     const float* __restrict__ w,
                                                     const float* __restrict__ bb,
                                                     __bf16* __restrict__ out) {
    int row = blockIdx.x, tid = threadIdx.x;
    const float* xr = x + (size_t)row * DIMX;
    float v0 = xr[tid], v1 = xr[tid + 256];
    __shared__ float s1[256], s2[256];
    s1[tid] = v0 + v1;
    s2[tid] = v0 * v0 + v1 * v1;
    __syncthreads();
    for (int st = 128; st > 0; st >>= 1) {
        if (tid < st) { s1[tid] += s1[tid + st]; s2[tid] += s2[tid + st]; }
        __syncthreads();
    }
    float m  = s1[0] * (1.f / DIMX);
    float vr = s2[0] * (1.f / DIMX) - m * m;
    float r  = rsqrtf(vr + EPSF);
    out[(size_t)row * DIMX + tid]       = (__bf16)((v0 - m) * r * w[tid] + bb[tid]);
    out[(size_t)row * DIMX + tid + 256] = (__bf16)((v1 - m) * r * w[tid + 256] + bb[tid + 256]);
}

// ---------------- post layernorm: in-place on d_out (BN,512) ----------------
__global__ __launch_bounds__(256) void ln_post_kernel(float* __restrict__ io,
                                                      const float* __restrict__ w,
                                                      const float* __restrict__ bb) {
    int row = blockIdx.x, tid = threadIdx.x;
    float* xr = io + (size_t)row * DIMX;
    float v0 = xr[tid], v1 = xr[tid + 256];
    __shared__ float s1[256], s2[256];
    s1[tid] = v0 + v1;
    s2[tid] = v0 * v0 + v1 * v1;
    __syncthreads();
    for (int st = 128; st > 0; st >>= 1) {
        if (tid < st) { s1[tid] += s1[tid + st]; s2[tid] += s2[tid + st]; }
        __syncthreads();
    }
    float m  = s1[0] * (1.f / DIMX);
    float vr = s2[0] * (1.f / DIMX) - m * m;
    float r  = rsqrtf(vr + EPSF);
    xr[tid]       = (v0 - m) * r * w[tid] + bb[tid];
    xr[tid + 256] = (v1 - m) * r * w[tid + 256] + bb[tid + 256];
}

// ---------------- generic bf16 WMMA GEMM ----------------
// C(M,N) = act(A(M,K) @ B(K,N) + bias);  optional fp32 accumulate into Cf
// tile 128x128, Kstep 32; 8 waves = 4(M)x2(N); wave = 32x64 out (2x4 wmma tiles)
__global__ __launch_bounds__(256) void gemm_bf16_kernel(
    const __bf16* __restrict__ A, const __bf16* __restrict__ Bm,
    const float* __restrict__ bias, float* __restrict__ Cf,
    __bf16* __restrict__ Cb, int M, int N, int K, int act, int accum) {
    __shared__ __align__(16) __bf16 sA[128][LDP];    // [row][k]
    __shared__ __align__(16) __bf16 sBt[128][LDP];   // [col][k]  (transposed)
    int tid = threadIdx.x, lane = tid & 31, wave = tid >> 5;
    int lr = lane & 15, lh = lane >> 4;
    int waveM = wave & 3, waveN = wave >> 2;
    int m0 = blockIdx.y * 128, n0 = blockIdx.x * 128;

    v8f acc[2][4];
#pragma unroll
    for (int mt = 0; mt < 2; mt++)
#pragma unroll
        for (int nt = 0; nt < 4; nt++) acc[mt][nt] = (v8f){};

    for (int k0 = 0; k0 < K; k0 += 32) {
        // A tile: vector load + vector store (16 bf16 per thread as 2x v8bf)
#pragma unroll
        for (int i = tid * 8; i < 128 * 32; i += 256 * 8) {
            int r = i >> 5, k = i & 31;
            int gr = m0 + r;
            v8bf va = (v8bf){};
            if (gr < M) va = *(const v8bf*)(A + (size_t)gr * K + k0 + k);
            *(v8bf*)&sA[r][k] = va;
        }
        // B tile: vector global load, transposed scalar LDS store
#pragma unroll
        for (int i = tid * 8; i < 32 * 128; i += 256 * 8) {
            int k = i >> 7, c = i & 127;
            v8bf vb = *(const v8bf*)(Bm + (size_t)(k0 + k) * N + n0 + c);
#pragma unroll
            for (int j = 0; j < 8; j++) sBt[c + j][k] = vb[j];
        }
        // prefetch next K-panel (global_prefetch_b8)
        if (k0 + 32 < K) {
            int i2 = tid * 8;
            int r = i2 >> 5, k = i2 & 31;
            int gr = m0 + r;
            if (gr < M) __builtin_prefetch(A + (size_t)gr * K + k0 + 32 + k, 0, 1);
            int kk = i2 >> 7, c = i2 & 127;
            __builtin_prefetch(Bm + (size_t)(k0 + 32 + kk) * N + n0 + c, 0, 1);
        }
        __syncthreads();
        v16bf afr[2], bfr[4];
#pragma unroll
        for (int mt = 0; mt < 2; mt++)
            afr[mt] = load_afrag(&sA[waveM * 32 + mt * 16 + lr][0], lh);
#pragma unroll
        for (int nt = 0; nt < 4; nt++)
            bfr[nt] = load_bfrag(&sBt[waveN * 64 + nt * 16 + lr][0], lh);
#pragma unroll
        for (int mt = 0; mt < 2; mt++)
#pragma unroll
            for (int nt = 0; nt < 4; nt++)
                acc[mt][nt] = __builtin_amdgcn_wmma_f32_16x16x32_bf16(
                    false, afr[mt], false, bfr[nt], (short)0, acc[mt][nt], false, false);
        __syncthreads();
    }

#pragma unroll
    for (int mt = 0; mt < 2; mt++)
#pragma unroll
        for (int nt = 0; nt < 4; nt++)
#pragma unroll
            for (int vv = 0; vv < 8; vv++) {
                int row = m0 + waveM * 32 + mt * 16 + lh * 8 + vv;
                int col = n0 + waveN * 64 + nt * 16 + lr;
                if (row < M) {
                    float val = acc[mt][nt][vv];
                    if (bias) val += bias[col];
                    if (act == 1) val = siluf(val);
                    size_t idx = (size_t)row * N + col;
                    if (Cf) {
                        if (accum) val += Cf[idx];
                        Cf[idx] = val;
                    }
                    if (Cb) Cb[idx] = (__bf16)val;
                }
            }
}

// ---------------- head split (gamma/beta) + RoPE ----------------
__global__ __launch_bounds__(128) void rope_heads_kernel(
    const __bf16* __restrict__ qk, const float* __restrict__ gamma,
    const float* __restrict__ beta, __bf16* __restrict__ qq, __bf16* __restrict__ lq,
    __bf16* __restrict__ qkk, __bf16* __restrict__ lk) {
    int t = blockIdx.x, d = threadIdx.x;
    int pos = t % N_SEQ;
    __shared__ float s[4][QKD];
    float qv = (float)qk[(size_t)t * QKD + d];
#pragma unroll
    for (int h = 0; h < 4; h++) s[h][d] = qv * gamma[h * QKD + d] + beta[h * QKD + d];
    __syncthreads();
    const int half = QKD / 2;
    int j = (d < half) ? d : d - half;
    float freq = __expf(-(float)j * (9.210340371976184f / (float)half));
    float ang = (float)pos * freq;
    float sn = sinf(ang), cs = cosf(ang);
    __bf16* outs[4] = {qq, lq, qkk, lk};
#pragma unroll
    for (int h = 0; h < 4; h++) {
        float x1 = s[h][j], x2 = s[h][j + half];
        float val = (d < half) ? (x1 * cs - x2 * sn) : (x2 * cs + x1 * sn);
        outs[h][(size_t)t * QKD + d] = (__bf16)val;
    }
}

// ---------------- quad chunk attention: scores VALU, attn@V via WMMA ----------
__global__ __launch_bounds__(256) void quad_attn_kernel(
    const __bf16* __restrict__ qq, const __bf16* __restrict__ qkk,
    const __bf16* __restrict__ vv, const int* __restrict__ mask,
    const float* __restrict__ pos_emb, float* __restrict__ mix) {
    int g = blockIdx.x, b = blockIdx.y;
    int tid = threadIdx.x;
    int base_tok = b * N_SEQ + g * G_SZ;
    __shared__ __align__(16) __bf16 sQ[32][QKD];
    __shared__ __align__(16) __bf16 sK[32][QKD];
    __shared__ __align__(16) __bf16 sAttn[32][LDP];  // [query][key]
    __shared__ __align__(16) __bf16 sVt[128][LDP];   // [col][token] (transposed)

    for (int i = tid * 8; i < 32 * QKD; i += 256 * 8) {
        int r = i >> 7, c = i & 127;
        v8bf zq = (v8bf){}, zk = (v8bf){};
        if (r < G_SZ) {
            zq = *(const v8bf*)(qq + (size_t)(base_tok + r) * QKD + c);
            zk = *(const v8bf*)(qkk + (size_t)(base_tok + r) * QKD + c);
        }
        *(v8bf*)&sQ[r][c] = zq;
        *(v8bf*)&sK[r][c] = zk;
    }
    __syncthreads();
    for (int i = tid; i < 32 * 32; i += 256) {
        int r = i >> 5, c = i & 31;
        float a = 0.f;
        if (r < G_SZ && c < G_SZ && mask[b * N_SEQ + g * G_SZ + c] != 0) {
            float s0 = 0.f;
            for (int d0 = 0; d0 < QKD; d0++) s0 += (float)sQ[r][d0] * (float)sK[c][d0];
            int rel = r - c; if (rel < 0) rel = -rel; if (rel > MAXREL) rel = MAXREL;
            float sc = s0 * (1.f / (float)G_SZ) + pos_emb[rel];
            sc = fmaxf(sc, 0.f);
            a = sc * sc;
        }
        sAttn[r][c] = (__bf16)a;
    }
    __syncthreads();

    int lane = tid & 31, wave = tid >> 5;
    int lr = lane & 15, lh = lane >> 4;
    int waveM = wave & 1, waveN = wave >> 1;   // 2 M tiles x 4 col groups of 32
    v16bf afr = load_afrag(&sAttn[waveM * 16 + lr][0], lh);
    for (int p = 0; p < 8; p++) {
        for (int i = tid * 8; i < 32 * 128; i += 256 * 8) {
            int r = i >> 7, c = i & 127;
            v8bf z = (v8bf){};
            if (r < G_SZ) z = *(const v8bf*)(vv + (size_t)(base_tok + r) * HIDX + p * 128 + c);
#pragma unroll
            for (int j = 0; j < 8; j++) sVt[c + j][r] = z[j];
        }
        __syncthreads();
#pragma unroll
        for (int nt = 0; nt < 2; nt++) {
            v16bf bfr = load_bfrag(&sVt[waveN * 32 + nt * 16 + lr][0], lh);
            v8f acc = (v8f){};
            acc = __builtin_amdgcn_wmma_f32_16x16x32_bf16(
                false, afr, false, bfr, (short)0, acc, false, false);
#pragma unroll
            for (int vv2 = 0; vv2 < 8; vv2++) {
                int row = waveM * 16 + lh * 8 + vv2;
                if (row < G_SZ) {
                    int col = p * 128 + waveN * 32 + nt * 16 + lr;
                    mix[(size_t)(base_tok + row) * HIDX + col] = acc[vv2];
                }
            }
        }
        __syncthreads();
    }
}

// ---------------- lin_kv split-K: (b,128,1024) += lin_k^T @ v / N --------------
__global__ __launch_bounds__(256) void linkv_kernel(
    const __bf16* __restrict__ lk, const __bf16* __restrict__ vv,
    float* __restrict__ linkv) {
    int eblk = blockIdx.x, split = blockIdx.y, b = blockIdx.z;
    int tid = threadIdx.x, lane = tid & 31, wave = tid >> 5;
    int lr = lane & 15, lh = lane >> 4;
    int waveM = wave & 3, waveN = wave >> 2;
    const int tokensPer = (N_SEQ + KV_SPLITS - 1) / KV_SPLITS;
    int t0 = split * tokensPer;
    int t1 = min(t0 + tokensPer, N_SEQ);
    __shared__ __align__(16) __bf16 sKt[128][LDP];   // [d][token]
    __shared__ __align__(16) __bf16 sVt[128][LDP];   // [e][token]
    v8f acc[2][4];
#pragma unroll
    for (int mt = 0; mt < 2; mt++)
#pragma unroll
        for (int nt = 0; nt < 4; nt++) acc[mt][nt] = (v8f){};

    for (int ts = t0; ts < t1; ts += 32) {
        for (int i = tid * 8; i < 32 * 128; i += 256 * 8) {
            int r = i >> 7, c = i & 127;
            int tok = ts + r;
            v8bf zk = (v8bf){}, zv = (v8bf){};
            if (tok < t1) {
                zk = *(const v8bf*)(lk + (size_t)(b * N_SEQ + tok) * QKD + c);
                zv = *(const v8bf*)(vv + (size_t)(b * N_SEQ + tok) * HIDX + eblk * 128 + c);
            }
#pragma unroll
            for (int j = 0; j < 8; j++) { sKt[c + j][r] = zk[j]; sVt[c + j][r] = zv[j]; }
        }
        __syncthreads();
        v16bf afr[2], bfr[4];
#pragma unroll
        for (int mt = 0; mt < 2; mt++)
            afr[mt] = load_afrag(&sKt[waveM * 32 + mt * 16 + lr][0], lh);
#pragma unroll
        for (int nt = 0; nt < 4; nt++)
            bfr[nt] = load_bfrag(&sVt[waveN * 64 + nt * 16 + lr][0], lh);
#pragma unroll
        for (int mt = 0; mt < 2; mt++)
#pragma unroll
            for (int nt = 0; nt < 4; nt++)
                acc[mt][nt] = __builtin_amdgcn_wmma_f32_16x16x32_bf16(
                    false, afr[mt], false, bfr[nt], (short)0, acc[mt][nt], false, false);
        __syncthreads();
    }
    const float invN = 1.f / (float)N_SEQ;
#pragma unroll
    for (int mt = 0; mt < 2; mt++)
#pragma unroll
        for (int nt = 0; nt < 4; nt++)
#pragma unroll
            for (int vv2 = 0; vv2 < 8; vv2++) {
                int d = waveM * 32 + mt * 16 + lh * 8 + vv2;
                int e = eblk * 128 + waveN * 64 + nt * 16 + lr;
                atomicAdd(&linkv[((size_t)b * QKD + d) * HIDX + e], acc[mt][nt][vv2] * invN);
            }
}

// ---------------- gating: y = bf16(gate * mix) ----------------
__global__ __launch_bounds__(256) void gate_mul_kernel(
    const __bf16* __restrict__ gate, const float* __restrict__ mix,
    __bf16* __restrict__ y, int n) {
    for (int i = blockIdx.x * blockDim.x + threadIdx.x; i < n; i += gridDim.x * blockDim.x)
        y[i] = (__bf16)((float)gate[i] * mix[i]);
}

// ======================================================================
extern "C" void kernel_launch(void* const* d_in, const int* in_sizes, int n_in,
                              void* d_out, int out_size, void* d_ws, size_t ws_size,
                              hipStream_t stream) {
    const float* x      = (const float*)d_in[0];
    const int*   mask   = (const int*)d_in[1];
    const float* pre_w  = (const float*)d_in[2];
    const float* pre_b  = (const float*)d_in[3];
    const float* qk_w   = (const float*)d_in[4];
    const float* qk_b   = (const float*)d_in[5];
    const float* v_w    = (const float*)d_in[6];
    const float* v_b    = (const float*)d_in[7];
    const float* gate_w = (const float*)d_in[8];
    const float* gate_b = (const float*)d_in[9];
    const float* out_w  = (const float*)d_in[10];
    const float* out_b  = (const float*)d_in[11];
    const float* post_w = (const float*)d_in[12];
    const float* post_b = (const float*)d_in[13];
    const float* gamma  = (const float*)d_in[14];
    const float* beta   = (const float*)d_in[15];
    const float* pose   = (const float*)d_in[16];
    float* out = (float*)d_out;

    char* wp = (char*)d_ws;
    auto alloc = [&](size_t bytes) {
        void* p = (void*)wp;
        wp += (bytes + 255) & ~(size_t)255;
        return p;
    };
    __bf16* xn_bf   = (__bf16*)alloc((size_t)BN * DIMX * 2);
    __bf16* wqk_bf  = (__bf16*)alloc((size_t)DIMX * QKD * 2);
    __bf16* wv_bf   = (__bf16*)alloc((size_t)DIMX * HIDX * 2);
    __bf16* wg_bf   = (__bf16*)alloc((size_t)DIMX * HIDX * 2);
    __bf16* wo_bf   = (__bf16*)alloc((size_t)HIDX * DIMX * 2);
    __bf16* qk_bf   = (__bf16*)alloc((size_t)BN * QKD * 2);
    __bf16* qq_bf   = (__bf16*)alloc((size_t)BN * QKD * 2);
    __bf16* lq_bf   = (__bf16*)alloc((size_t)BN * QKD * 2);
    __bf16* qkk_bf  = (__bf16*)alloc((size_t)BN * QKD * 2);
    __bf16* lk_bf   = (__bf16*)alloc((size_t)BN * QKD * 2);
    __bf16* v_bf    = (__bf16*)alloc((size_t)BN * HIDX * 2);
    __bf16* gate_bf = (__bf16*)alloc((size_t)BN * HIDX * 2);
    float*  mix_f   = (float*)alloc((size_t)BN * HIDX * 4);
    float*  linkv_f = (float*)alloc((size_t)B_SZ * QKD * HIDX * 4);
    __bf16* linkv_bf= (__bf16*)alloc((size_t)B_SZ * QKD * HIDX * 2);
    __bf16* y_bf    = (__bf16*)alloc((size_t)BN * HIDX * 2);

    // 1) weight casts
    f2bf_kernel<<<256, 256, 0, stream>>>(qk_w,   wqk_bf, DIMX * QKD);
    f2bf_kernel<<<512, 256, 0, stream>>>(v_w,    wv_bf,  DIMX * HIDX);
    f2bf_kernel<<<512, 256, 0, stream>>>(gate_w, wg_bf,  DIMX * HIDX);
    f2bf_kernel<<<512, 256, 0, stream>>>(out_w,  wo_bf,  HIDX * DIMX);

    // 2) pre layernorm
    ln_pre_kernel<<<BN, 256, 0, stream>>>(x, pre_w, pre_b, xn_bf);

    // 3) qk / v / gate projections (bias + silu fused), WMMA
    gemm_bf16_kernel<<<dim3(QKD / 128, (BN + 127) / 128), 256, 0, stream>>>(
        xn_bf, wqk_bf, qk_b, nullptr, qk_bf, BN, QKD, DIMX, 1, 0);
    gemm_bf16_kernel<<<dim3(HIDX / 128, (BN + 127) / 128), 256, 0, stream>>>(
        xn_bf, wv_bf, v_b, nullptr, v_bf, BN, HIDX, DIMX, 1, 0);
    gemm_bf16_kernel<<<dim3(HIDX / 128, (BN + 127) / 128), 256, 0, stream>>>(
        xn_bf, wg_bf, gate_b, nullptr, gate_bf, BN, HIDX, DIMX, 1, 0);

    // 4) head split + rope
    rope_heads_kernel<<<BN, 128, 0, stream>>>(qk_bf, gamma, beta,
                                              qq_bf, lq_bf, qkk_bf, lk_bf);

    // 5) quad chunk attention -> mix (fp32), covers every token exactly once
    quad_attn_kernel<<<dim3(GCHUNK, B_SZ), 256, 0, stream>>>(
        qq_bf, qkk_bf, v_bf, mask, pose, mix_f);

    // 6) lin_kv (split-K WMMA, atomic fp32 accumulate), then cast bf16
    hipMemsetAsync(linkv_f, 0, (size_t)B_SZ * QKD * HIDX * 4, stream);
    linkv_kernel<<<dim3(HIDX / 128, KV_SPLITS, B_SZ), 256, 0, stream>>>(
        lk_bf, v_bf, linkv_f);
    f2bf_kernel<<<256, 256, 0, stream>>>(linkv_f, linkv_bf, B_SZ * QKD * HIDX);

    // 7) lin_out = lin_q @ lin_kv, accumulated into mix (per batch)
    for (int b = 0; b < B_SZ; b++) {
        gemm_bf16_kernel<<<dim3(HIDX / 128, (N_SEQ + 127) / 128), 256, 0, stream>>>(
            lq_bf + (size_t)b * N_SEQ * QKD,
            linkv_bf + (size_t)b * QKD * HIDX,
            nullptr,
            mix_f + (size_t)b * N_SEQ * HIDX,
            nullptr, N_SEQ, HIDX, QKD, 0, 1);
    }

    // 8) gating
    gate_mul_kernel<<<4096, 256, 0, stream>>>(gate_bf, mix_f, y_bf, BN * HIDX);

    // 9) output projection + bias -> d_out, then post layernorm in place
    gemm_bf16_kernel<<<dim3(DIMX / 128, (BN + 127) / 128), 256, 0, stream>>>(
        y_bf, wo_bf, out_b, out, nullptr, BN, DIMX, HIDX, 0, 0);
    ln_post_kernel<<<BN, 256, 0, stream>>>(out, post_w, post_b);
}
